// AugmentedOUDFollmerSDESTL_22454089024141
// MI455X (gfx1250) — matrix operand order repaired and checked by reading, not compiled, not run on hardware
//
#include <hip/hip_runtime.h>

typedef _Float16 v16h __attribute__((ext_vector_type(16)));
typedef _Float16 v8h  __attribute__((ext_vector_type(8)));
typedef float    v8f  __attribute__((ext_vector_type(8)));

#define DIM     128
#define HIDDEN  512
#define NSTEPS  20
#define BATCH   16384
#define OUTD    131           // DIM + 3 augmented columns
#define WAVES   4             // waves per block
#define ROWS_PER_BLOCK (WAVES * 16)

// Hardware tanh (CDNA5 TRANS op v_tanh_f32) with exp2-based fallback.
#if __has_builtin(__builtin_amdgcn_tanhf)
__device__ __forceinline__ float fast_tanh(float x) {
  return __builtin_amdgcn_tanhf(x);
}
#else
__device__ __forceinline__ float fast_tanh(float x) {
  float e = __builtin_amdgcn_exp2f(x * 2.8853900817779268f);
  return 1.0f - 2.0f * __builtin_amdgcn_rcpf(e + 1.0f);
}
#endif

// ---------------------------------------------------------------------------
// A-fragment (16x32, f16): lane L<16 holds row L, K = {k0..k0+15}
// with k0 = kTile*32; lanes >=16 hold the same rows with K shifted by 16.
// Tile is row-major in LDS, stride 128 f16 -> two ds_load_b128.
// ---------------------------------------------------------------------------
__device__ __forceinline__ v16h load_a16(const _Float16* __restrict__ tile,
                                         int lane, int kTile) {
  const bool hi = lane >= 16;
  const int row = lane & 15;
  const int k0  = kTile * 32 + (hi ? 16 : 0);
  v8h p = *(const v8h*)(tile + row * 128 + k0);
  v8h q = *(const v8h*)(tile + row * 128 + k0 + 8);
  v16h a;
#pragma unroll
  for (int i = 0; i < 8; i++) { a[i] = p[i]; a[i + 8] = q[i]; }
  return a;
}

// B-fragment from a precomputed per-lane base + compile-time element offset,
// so the DS 16-bit immediate-offset field is used (no VALU address adds).
__device__ __forceinline__ v16h load_b_off(const _Float16* __restrict__ base,
                                           int off) {
  v8h x = *(const v8h*)(base + off);
  v8h y = *(const v8h*)(base + off + 8);
  v16h b;
#pragma unroll
  for (int i = 0; i < 8; i++) { b[i] = x[i]; b[i + 8] = y[i]; }
  return b;
}

// ---------------------------------------------------------------------------
// Prep: transpose + f16-convert weights into workspace.
//   w1t[n][k] = W1[k][n]   (512 x 128)   (k < 128; k==128 handled via w1last)
//   w2t[d][k] = W2[k][d]   (128 x 512)
//   w1last[n] = W1[128][n] (t-row of W1, f32)
// ---------------------------------------------------------------------------
__global__ void prep_kernel(const float* __restrict__ W1,
                            const float* __restrict__ W2,
                            _Float16* __restrict__ w1t,
                            _Float16* __restrict__ w2t,
                            float* __restrict__ w1last) {
  int i0 = blockIdx.x * blockDim.x + threadIdx.x;
  int stride = gridDim.x * blockDim.x;
  for (int i = i0; i < HIDDEN * DIM; i += stride) {
    int n = i >> 7, k = i & 127;
    w1t[i] = (_Float16)W1[k * HIDDEN + n];
  }
  for (int i = i0; i < DIM * HIDDEN; i += stride) {
    int d = i >> 9, k = i & 511;
    w2t[i] = (_Float16)W2[k * DIM + d];
  }
  for (int i = i0; i < HIDDEN; i += stride)
    w1last[i] = W1[DIM * HIDDEN + i];
}

__global__ void ts_kernel(const float* __restrict__ ts, float* __restrict__ out) {
  int i = threadIdx.x;
  if (i <= NSTEPS)
    out[(size_t)(NSTEPS + 1) * BATCH * OUTD + i] = ts[i];
}

// ---------------------------------------------------------------------------
// Main persistent SDE kernel. Each wave integrates 16 batch rows for all 20
// steps; weights live in LDS for the whole kernel. f32 state is carried in
// the trajectory rows (L2-resident) + an f16 LDS staging tile for WMMA.
// ---------------------------------------------------------------------------
__global__ void __launch_bounds__(WAVES * 32, 1)
sde_kernel(const float* __restrict__ y0,  const float* __restrict__ b1,
           const float* __restrict__ b2,  const float* __restrict__ noises,
           const float* __restrict__ ts,  const _Float16* __restrict__ w1tG,
           const _Float16* __restrict__ w2tG, const float* __restrict__ w1last,
           float* __restrict__ out) {
  extern __shared__ char smemRaw[];
  _Float16* w1t = (_Float16*)smemRaw;                  // 512*128 f16 = 128 KB
  _Float16* w2t = w1t + HIDDEN * DIM;                  // 128*512 f16 = 128 KB
  float*    c1  = (float*)(w2t + DIM * HIDDEN);        // 512 f32
  _Float16* yShB = (_Float16*)(c1 + HIDDEN);           // WAVES * 16*128 f16
  _Float16* hShB = yShB + WAVES * 16 * 128;            // WAVES * 16*128 f16

  const int tid   = threadIdx.x;
  const int lane  = tid & 31;
  const int wave  = tid >> 5;
  const bool hi   = lane >= 16;
  const int nIdx  = lane & 15;
  const int mbase = hi ? 8 : 0;

  // Stage both weight matrices into LDS once (b128 copies).
  {
    const int4* s1 = (const int4*)w1tG;  int4* d1 = (int4*)w1t;
    for (int i = tid; i < (HIDDEN * DIM * 2) / 16; i += WAVES * 32) d1[i] = s1[i];
    const int4* s2 = (const int4*)w2tG;  int4* d2 = (int4*)w2t;
    for (int i = tid; i < (DIM * HIDDEN * 2) / 16; i += WAVES * 32) d2[i] = s2[i];
  }
  __syncthreads();

  _Float16* ySh = yShB + wave * 16 * 128;
  _Float16* hSh = hShB + wave * 16 * 128;
  const int rowBase = (blockIdx.x * WAVES + wave) * 16;

  // Init: y0 -> step-0 trajectory (f32) and ySh staging tile (f16).
#pragma unroll
  for (int nt = 0; nt < 8; nt++) {
    int col = nt * 16 + nIdx;
#pragma unroll
    for (int m = 0; m < 8; m++) {
      float v = y0[(size_t)(rowBase + mbase + m) * DIM + col];
      out[(size_t)(rowBase + mbase + m) * OUTD + col] = v;
      ySh[(mbase + m) * 128 + col] = (_Float16)v;
    }
  }
  if (nIdx < 8) {
    size_t r = (size_t)(rowBase + mbase + nIdx) * OUTD;
    out[r + 128] = 0.f; out[r + 129] = 0.f; out[r + 130] = 0.f;
  }

  float augU[8], augE[8];
#pragma unroll
  for (int m = 0; m < 8; m++) { augU[m] = 0.f; augE[m] = 0.f; }

  for (int s = 0; s < NSTEPS; s++) {
    const float t  = ts[s];
    const float dt = ts[s + 1] - t;
    const float sq = sqrtf(dt);

    const float* noiseS = noises + (size_t)s * BATCH * DIM;
    const float* yOldS  = out + (size_t)s * BATCH * OUTD;
    float*       outS   = out + (size_t)(s + 1) * BATCH * OUTD;

    // Early prefetch of this step's noise + previous-state tiles (L2/WGP$).
    {
      const float* pfN = noiseS + (size_t)rowBase * DIM;   // 8 KB tile
      __builtin_prefetch(pfN + lane * 64, 0, 0);
      __builtin_prefetch(pfN + lane * 64 + 32, 0, 0);
      const float* pfY = yOldS + (size_t)rowBase * OUTD;   // ~8.4 KB tile
      __builtin_prefetch(pfY + lane * 66, 0, 0);
      __builtin_prefetch(pfY + lane * 66 + 33, 0, 0);
    }

    // c1[n] = b1[n] + t * W1[128][n]  (bias + t-column folded into GEMM1 init)
    __syncthreads();
    for (int i = tid; i < HIDDEN; i += WAVES * 32)
      c1[i] = b1[i] + t * w1last[i];
    __syncthreads();

    // y A-fragments: shared by every GEMM1 tile this step -> load once.
    v16h aY[4];
#pragma unroll
    for (int kt = 0; kt < 4; kt++) aY[kt] = load_a16(ySh, lane, kt);

    // GEMM2 accumulators start at b2.
    v8f U[8];
#pragma unroll
    for (int nt = 0; nt < 8; nt++) {
      float bv = b2[nt * 16 + nIdx];
#pragma unroll
      for (int m = 0; m < 8; m++) U[nt][m] = bv;
    }

    for (int c = 0; c < 4; c++) {                 // hidden chunk of 128
      // Preload chunk biases so no DS wait sits in front of the WMMA chains.
      float cv[8];
#pragma unroll
      for (int nt = 0; nt < 8; nt++) cv[nt] = c1[c * 128 + nt * 16 + nIdx];

      // GEMM1: h_pre = c1 + y @ W1[:128] -> tanh -> hSh (f16).
      // Two interleaved N-tiles + double-buffered B: WMMA of one chain hides
      // the ds_load latency of the other.
      const _Float16* w1base = w1t + (c * 128 + nIdx) * 128 + (hi ? 16 : 0);
#pragma unroll
      for (int ntp = 0; ntp < 4; ntp++) {
        const int nt0 = ntp * 2, nt1 = nt0 + 1;
        v8f acc0, acc1;
#pragma unroll
        for (int m = 0; m < 8; m++) { acc0[m] = cv[nt0]; acc1[m] = cv[nt1]; }
        v16h b0 = load_b_off(w1base, nt0 * 2048);
        v16h b1v = load_b_off(w1base, nt1 * 2048);
#pragma unroll
        for (int kt = 0; kt < 4; kt++) {
          v16h c0 = b0, c1v = b1v;
          if (kt < 3) {
            b0  = load_b_off(w1base, nt0 * 2048 + (kt + 1) * 32);
            b1v = load_b_off(w1base, nt1 * 2048 + (kt + 1) * 32);
          }
          acc0 = __builtin_amdgcn_wmma_f32_16x16x32_f16(
              false, aY[kt], false, c0, (short)0, acc0, false, false);
          acc1 = __builtin_amdgcn_wmma_f32_16x16x32_f16(
              false, aY[kt], false, c1v, (short)0, acc1, false, false);
        }
        const int col0 = nt0 * 16 + nIdx, col1 = nt1 * 16 + nIdx;
#pragma unroll
        for (int m = 0; m < 8; m++) {
          hSh[(mbase + m) * 128 + col0] = (_Float16)fast_tanh(acc0[m]);
          hSh[(mbase + m) * 128 + col1] = (_Float16)fast_tanh(acc1[m]);
        }
      }

      // h A-fragments for this chunk: load once, reuse across 8 N-tiles.
      v16h aH[4];
#pragma unroll
      for (int kt = 0; kt < 4; kt++) aH[kt] = load_a16(hSh, lane, kt);

      // GEMM2 partial: U += h_chunk @ W2[chunk, :], same pairing scheme.
      // Two bases keep nt-offsets under the 16-bit DS offset limit.
      const _Float16* w2b0 = w2t + nIdx * 512 + c * 128 + (hi ? 16 : 0);
      const _Float16* w2b1 = w2b0 + 4 * 8192;
#pragma unroll
      for (int ntp = 0; ntp < 4; ntp++) {
        const int nt0 = ntp * 2, nt1 = nt0 + 1;
        const _Float16* wb = (ntp < 2) ? w2b0 : w2b1;
        const int o0 = (nt0 & 3) * 8192, o1 = (nt1 & 3) * 8192;
        v16h b0 = load_b_off(wb, o0);
        v16h b1v = load_b_off(wb, o1);
#pragma unroll
        for (int kt = 0; kt < 4; kt++) {
          v16h c0 = b0, c1v = b1v;
          if (kt < 3) {
            b0  = load_b_off(wb, o0 + (kt + 1) * 32);
            b1v = load_b_off(wb, o1 + (kt + 1) * 32);
          }
          U[nt0] = __builtin_amdgcn_wmma_f32_16x16x32_f16(
              false, aH[kt], false, c0, (short)0, U[nt0], false, false);
          U[nt1] = __builtin_amdgcn_wmma_f32_16x16x32_f16(
              false, aH[kt], false, c1v, (short)0, U[nt1], false, false);
        }
      }
    }

    // Euler-Maruyama update + per-row reductions + stores.
    // y_old re-read in f32 from the trajectory row written last step.
    float pUN[8], pUU[8];
#pragma unroll
    for (int m = 0; m < 8; m++) { pUN[m] = 0.f; pUU[m] = 0.f; }

#pragma unroll
    for (int nt = 0; nt < 8; nt++) {
      int col = nt * 16 + nIdx;
#pragma unroll
      for (int m = 0; m < 8; m++) {
        int row = rowBase + mbase + m;
        float nz  = noiseS[(size_t)row * DIM + col];
        float yo  = yOldS[(size_t)row * OUTD + col];
        float uvv = U[nt][m];
        float yn  = yo + (uvv - yo) * dt + sq * nz;   // ALPHA = SIGMA = 1
        outS[(size_t)row * OUTD + col] = yn;
        ySh[(mbase + m) * 128 + col] = (_Float16)yn;  // stage for next step
        pUN[m] += uvv * nz;
        pUU[m] += uvv * uvv;
      }
    }
    // Butterfly reduce over the 16 lanes sharing each row group.
#pragma unroll
    for (int m = 0; m < 8; m++) {
      float a = pUN[m], e = pUU[m];
#pragma unroll
      for (int off = 1; off < 16; off <<= 1) {
        a += __shfl_xor(a, off, 32);
        e += __shfl_xor(e, off, 32);
      }
      augU[m] += a * sq;          // int u/sigma . dW  (sigma = 1)
      augE[m] += 0.5f * e * dt;   // int |u|^2 / 2 dt
    }
    if (nIdx < 8) {
      size_t r = (size_t)(rowBase + mbase + nIdx) * OUTD;
      outS[r + 128] = augU[nIdx];
      outS[r + 129] = augU[nIdx];   // stop_gradient copy == same in fwd
      outS[r + 130] = augE[nIdx];
    }
  }
}

extern "C" void kernel_launch(void* const* d_in, const int* in_sizes, int n_in,
                              void* d_out, int out_size, void* d_ws, size_t ws_size,
                              hipStream_t stream) {
  const float* y0     = (const float*)d_in[0];
  const float* W1     = (const float*)d_in[1];
  const float* b1     = (const float*)d_in[2];
  const float* W2     = (const float*)d_in[3];
  const float* b2     = (const float*)d_in[4];
  const float* noises = (const float*)d_in[5];
  const float* ts     = (const float*)d_in[6];
  float* out = (float*)d_out;

  _Float16* w1t    = (_Float16*)d_ws;              // 512*128 f16
  _Float16* w2t    = w1t + HIDDEN * DIM;           // 128*512 f16
  float*    w1last = (float*)(w2t + DIM * HIDDEN); // 512 f32

  prep_kernel<<<64, 256, 0, stream>>>(W1, W2, w1t, w2t, w1last);
  ts_kernel<<<1, 32, 0, stream>>>(ts, out);

  const size_t shmem =
      (size_t)(HIDDEN * DIM + DIM * HIDDEN + 2 * WAVES * 16 * 128) * sizeof(_Float16)
      + HIDDEN * sizeof(float);                    // = 296,960 B (< 320 KB/WGP)
  (void)hipFuncSetAttribute((const void*)sde_kernel,
                            hipFuncAttributeMaxDynamicSharedMemorySize,
                            (int)shmem);
  sde_kernel<<<BATCH / ROWS_PER_BLOCK, WAVES * 32, shmem, stream>>>(
      y0, b1, b2, noises, ts, w1t, w2t, w1last, out);
}
